// GraphNetBlock_60894046322879
// MI455X (gfx1250) — matrix operand order repaired and checked by reading, not compiled, not run on hardware
//
#include <hip/hip_runtime.h>

typedef __attribute__((ext_vector_type(16))) __bf16 v16bf;
typedef __attribute__((ext_vector_type(8)))  float  v8f;

union FragBF {
    uint4 q[2];
    v16bf v;
};

__device__ __forceinline__ unsigned short f2bf(float f) {
    union { float f; unsigned u; } x;
    x.f = f;
    unsigned u = x.u;
    // round-to-nearest-even f32 -> bf16
    unsigned r = (u + 0x7FFFu + ((u >> 16) & 1u)) >> 16;
    return (unsigned short)r;
}

// ---------------------------------------------------------------------------
// Kernel 1: zero the aggregation buffer in workspace
// ---------------------------------------------------------------------------
__global__ void zero_f4_kernel(float4* __restrict__ p, int n4) {
    int i = blockIdx.x * blockDim.x + threadIdx.x;
    if (i < n4) {
        float4 z; z.x = 0.f; z.y = 0.f; z.z = 0.f; z.w = 0.f;
        p[i] = z;
    }
}

// ---------------------------------------------------------------------------
// Kernel 2: agg[recv[e]][d] += edge_features[e][d]  (L2-resident atomics)
// ---------------------------------------------------------------------------
__global__ void scatter_add_kernel(const float* __restrict__ ef,
                                   const int* __restrict__ recv,
                                   float* __restrict__ agg, int total) {
    int idx = blockIdx.x * blockDim.x + threadIdx.x;
    if (idx >= total) return;
    int e = idx >> 7;
    int d = idx & 127;
    atomicAdd(agg + (size_t)recv[e] * 128 + d, ef[idx]);
}

// ---------------------------------------------------------------------------
// Fused MLP (2 layers, no activation) + LayerNorm over last dim (=128).
// MODE 0: node — input row r = concat(node_features[r], agg[r])          (KIN=256)
// MODE 1: edge — input row r = concat(nodes[snd[r]], nodes[rcv[r]], ef[r]) (KIN=384)
// Block: 256 threads = 8 wave32; tile = 32 rows x 128 cols of output.
// ---------------------------------------------------------------------------
template<int KIN> struct MLPCfg {
    static constexpr int    LDW1 = KIN + 8;   // bf16 elems per W1^T row (pad 16B)
    static constexpr int    LDW2 = 128 + 8;
    static constexpr int    LDA  = KIN + 8;   // bf16 input tile row
    static constexpr int    LDH1 = 128 + 8;   // bf16 h1 row
    static constexpr int    LDH2 = 128 + 4;   // f32 h2 row
    static constexpr size_t off_w1  = 0;
    static constexpr size_t sz_w1   = (size_t)128 * LDW1 * 2;
    static constexpr size_t off_w2  = off_w1 + sz_w1;
    static constexpr size_t sz_w2   = (size_t)128 * LDW2 * 2;
    static constexpr size_t off_a   = off_w2 + sz_w2;           // aliased with h2
    static constexpr size_t sz_a0   = (size_t)32 * LDA  * 2;
    static constexpr size_t sz_h2   = (size_t)32 * LDH2 * 4;
    static constexpr size_t sz_a    = (sz_a0 > sz_h2) ? sz_a0 : sz_h2;
    static constexpr size_t off_h1  = off_a + sz_a;
    static constexpr size_t sz_h1   = (size_t)32 * LDH1 * 2;
    static constexpr size_t off_bias= off_h1 + sz_h1;
    static constexpr size_t total   = off_bias + (size_t)4 * 128 * 4;
};

template<int KIN, int MODE>
__global__ __launch_bounds__(256, 1)
void mlp_ln_kernel(const float* __restrict__ in0,   // MODE0: node_features ; MODE1: new_nodes
                   const float* __restrict__ in1,   // MODE0: agg           ; MODE1: unused
                   const float* __restrict__ in2,   // MODE1: edge_features
                   const int*   __restrict__ snd,
                   const int*   __restrict__ rcv,
                   const float* __restrict__ W1, const float* __restrict__ b1,
                   const float* __restrict__ W2, const float* __restrict__ b2,
                   const float* __restrict__ lng, const float* __restrict__ lnb,
                   float* __restrict__ out, int nrows, int ntiles) {
    typedef MLPCfg<KIN> C;
    extern __shared__ char smem[];
    unsigned short* sW1  = (unsigned short*)(smem + C::off_w1);
    unsigned short* sW2  = (unsigned short*)(smem + C::off_w2);
    unsigned short* sA   = (unsigned short*)(smem + C::off_a);
    float*          sH2  = (float*)        (smem + C::off_a);   // alias (A dead after L1)
    unsigned short* sH1  = (unsigned short*)(smem + C::off_h1);
    float*          sBia = (float*)        (smem + C::off_bias);

    const int tid  = threadIdx.x;
    const int w    = tid >> 5;
    const int lane = tid & 31;
    const int nloc = lane & 15;
    const int mhalf= (lane >> 4) & 1;
    const int n0   = w * 16;

    // ---- stage weights transposed (bf16) + biases, once per block ----
    for (int idx = tid; idx < KIN * 128; idx += 256) {
        int k = idx >> 7, n = idx & 127;
        sW1[n * C::LDW1 + k] = f2bf(W1[idx]);
    }
    for (int idx = tid; idx < 128 * 128; idx += 256) {
        int k = idx >> 7, n = idx & 127;
        sW2[n * C::LDW2 + k] = f2bf(W2[idx]);
    }
    if (tid < 128) {
        sBia[tid]       = b1[tid];
        sBia[128 + tid] = b2[tid];
        sBia[256 + tid] = lng[tid];
        sBia[384 + tid] = lnb[tid];
    }
    __syncthreads();

    for (int t = blockIdx.x; t < ntiles; t += gridDim.x) {
        const int r0 = t * 32;

        // ---- build 32 x KIN bf16 input tile ----
        constexpr int F4 = KIN / 4;
        for (int idx = tid; idx < 32 * F4; idx += 256) {
            int i = idx / F4, q = idx - i * F4;
            int r = r0 + i; if (r >= nrows) r = nrows - 1;
            int c = q * 4;
            const float* src;
            if (MODE == 0) {
                src = (c < 128) ? (in0 + (size_t)r * 128 + c)
                                : (in1 + (size_t)r * 128 + (c - 128));
            } else {
                if (c < 128)       src = in0 + (size_t)snd[r] * 128 + c;
                else if (c < 256)  src = in0 + (size_t)rcv[r] * 128 + (c - 128);
                else               src = in2 + (size_t)r * 128 + (c - 256);
            }
            float4 v = *(const float4*)src;
            unsigned short* dst = sA + (size_t)i * C::LDA + c;
            dst[0] = f2bf(v.x); dst[1] = f2bf(v.y);
            dst[2] = f2bf(v.z); dst[3] = f2bf(v.w);
        }
        __syncthreads();

        // ---- layer 1: [32,KIN] @ [KIN,128] ; wave w owns cols n0..n0+15 ----
        v8f acc0 = {}, acc1 = {};
        for (int kc = 0; kc < KIN; kc += 32) {
            FragBF B;
            const unsigned short* colp = sW1 + (size_t)(n0 + nloc) * C::LDW1 + kc + mhalf * 8;
            B.q[0] = *(const uint4*)(colp);
            B.q[1] = *(const uint4*)(colp + 16);
            FragBF A0, A1;
            const unsigned short* rp0 = sA + (size_t)nloc * C::LDA + kc + mhalf * 8;
            A0.q[0] = *(const uint4*)(rp0);
            A0.q[1] = *(const uint4*)(rp0 + 16);
            const unsigned short* rp1 = sA + (size_t)(16 + nloc) * C::LDA + kc + mhalf * 8;
            A1.q[0] = *(const uint4*)(rp1);
            A1.q[1] = *(const uint4*)(rp1 + 16);
            acc0 = __builtin_amdgcn_wmma_f32_16x16x32_bf16(false, A0.v, false, B.v, (short)0, acc0, false, false);
            acc1 = __builtin_amdgcn_wmma_f32_16x16x32_bf16(false, A1.v, false, B.v, (short)0, acc1, false, false);
        }
        {
            float bias1 = sBia[n0 + nloc];
            for (int v = 0; v < 8; ++v) {
                int m = v + mhalf * 8;   // D layout: VGPR v -> row v (+8 for lanes>=16)
                sH1[(size_t)m        * C::LDH1 + n0 + nloc] = f2bf(acc0[v] + bias1);
                sH1[(size_t)(16 + m) * C::LDH1 + n0 + nloc] = f2bf(acc1[v] + bias1);
            }
        }
        __syncthreads();

        // ---- layer 2: [32,128] @ [128,128] ----
        v8f c0 = {}, c1 = {};
        for (int kc = 0; kc < 128; kc += 32) {
            FragBF B;
            const unsigned short* colp = sW2 + (size_t)(n0 + nloc) * C::LDW2 + kc + mhalf * 8;
            B.q[0] = *(const uint4*)(colp);
            B.q[1] = *(const uint4*)(colp + 16);
            FragBF A0, A1;
            const unsigned short* rp0 = sH1 + (size_t)nloc * C::LDH1 + kc + mhalf * 8;
            A0.q[0] = *(const uint4*)(rp0);
            A0.q[1] = *(const uint4*)(rp0 + 16);
            const unsigned short* rp1 = sH1 + (size_t)(16 + nloc) * C::LDH1 + kc + mhalf * 8;
            A1.q[0] = *(const uint4*)(rp1);
            A1.q[1] = *(const uint4*)(rp1 + 16);
            c0 = __builtin_amdgcn_wmma_f32_16x16x32_bf16(false, A0.v, false, B.v, (short)0, c0, false, false);
            c1 = __builtin_amdgcn_wmma_f32_16x16x32_bf16(false, A1.v, false, B.v, (short)0, c1, false, false);
        }
        {
            float bias2 = sBia[128 + n0 + nloc];
            for (int v = 0; v < 8; ++v) {
                int m = v + mhalf * 8;
                sH2[(size_t)m        * C::LDH2 + n0 + nloc] = c0[v] + bias2;
                sH2[(size_t)(16 + m) * C::LDH2 + n0 + nloc] = c1[v] + bias2;
            }
        }
        __syncthreads();

        // ---- LayerNorm over 128 cols; 8 waves x 4 rows ----
        for (int rr = 0; rr < 4; ++rr) {
            int i = w * 4 + rr;
            int r = r0 + i;
            const float* hr = sH2 + (size_t)i * C::LDH2;
            float4 x = *(const float4*)(hr + lane * 4);
            float s  = x.x + x.y + x.z + x.w;
            float sq = x.x * x.x + x.y * x.y + x.z * x.z + x.w * x.w;
            for (int off = 16; off > 0; off >>= 1) {
                s  += __shfl_xor(s,  off, 32);
                sq += __shfl_xor(sq, off, 32);
            }
            float mean = s * (1.f / 128.f);
            float var  = sq * (1.f / 128.f) - mean * mean;
            float rstd = rsqrtf(var + 1e-5f);
            if (r < nrows) {
                int c = lane * 4;
                float4 o;
                o.x = (x.x - mean) * rstd * sBia[256 + c + 0] + sBia[384 + c + 0];
                o.y = (x.y - mean) * rstd * sBia[256 + c + 1] + sBia[384 + c + 1];
                o.z = (x.z - mean) * rstd * sBia[256 + c + 2] + sBia[384 + c + 2];
                o.w = (x.w - mean) * rstd * sBia[256 + c + 3] + sBia[384 + c + 3];
                *(float4*)(out + (size_t)r * 128 + c) = o;
            }
        }
        __syncthreads();   // protect sA/sH1 before next tile
    }
}

// ---------------------------------------------------------------------------
extern "C" void kernel_launch(void* const* d_in, const int* in_sizes, int n_in,
                              void* d_out, int out_size, void* d_ws, size_t ws_size,
                              hipStream_t stream) {
    const float* node_features = (const float*)d_in[0];
    const float* edge_features = (const float*)d_in[1];
    const int*   senders       = (const int*)d_in[2];
    const int*   receivers     = (const int*)d_in[3];
    const float* node_w1 = (const float*)d_in[4];
    const float* node_b1 = (const float*)d_in[5];
    const float* node_w2 = (const float*)d_in[6];
    const float* node_b2 = (const float*)d_in[7];
    const float* node_g  = (const float*)d_in[8];
    const float* node_bb = (const float*)d_in[9];
    const float* edge_w1 = (const float*)d_in[10];
    const float* edge_b1 = (const float*)d_in[11];
    const float* edge_w2 = (const float*)d_in[12];
    const float* edge_b2 = (const float*)d_in[13];
    const float* edge_g  = (const float*)d_in[14];
    const float* edge_bb = (const float*)d_in[15];

    const int N = in_sizes[0] / 128;
    const int E = in_sizes[1] / 128;

    float* agg       = (float*)d_ws;            // N*128 f32 scratch (L2-resident)
    float* new_nodes = (float*)d_out;           // N*128
    float* new_edges = (float*)d_out + (size_t)N * 128;  // E*128

    // 1) zero agg
    {
        int n4 = (N * 128) / 4;
        zero_f4_kernel<<<(n4 + 255) / 256, 256, 0, stream>>>((float4*)agg, n4);
    }
    // 2) segment-sum via f32 atomics
    {
        int total = E * 128;
        scatter_add_kernel<<<(total + 255) / 256, 256, 0, stream>>>(edge_features, receivers, agg, total);
    }
    // 3) node MLP + LN
    {
        size_t sh = MLPCfg<256>::total;
        (void)hipFuncSetAttribute((const void*)&mlp_ln_kernel<256, 0>,
                                  hipFuncAttributeMaxDynamicSharedMemorySize, (int)sh);
        int ntiles = (N + 31) / 32;
        int blocks = ntiles < 1024 ? ntiles : 1024;
        mlp_ln_kernel<256, 0><<<blocks, 256, sh, stream>>>(
            node_features, agg, nullptr, nullptr, nullptr,
            node_w1, node_b1, node_w2, node_b2, node_g, node_bb,
            new_nodes, N, ntiles);
    }
    // 4) edge MLP + LN (gathers new_nodes from d_out)
    {
        size_t sh = MLPCfg<384>::total;
        (void)hipFuncSetAttribute((const void*)&mlp_ln_kernel<384, 1>,
                                  hipFuncAttributeMaxDynamicSharedMemorySize, (int)sh);
        int ntiles = (E + 31) / 32;
        int blocks = ntiles < 2048 ? ntiles : 2048;
        mlp_ln_kernel<384, 1><<<blocks, 256, sh, stream>>>(
            new_nodes, nullptr, edge_features, senders, receivers,
            edge_w1, edge_b1, edge_w2, edge_b2, edge_g, edge_bb,
            new_edges, E, ntiles);
    }
}
